// NNBase_31894427140492
// MI455X (gfx1250) — matrix operand order repaired and checked by reading, not compiled, not run on hardware
//
#include <hip/hip_runtime.h>

typedef __attribute__((ext_vector_type(16))) __bf16 v16bf;
typedef __attribute__((ext_vector_type(8)))  float  v8f;

#define T_STEPS 512
#define NBATCH  512
#define IDIM    128
#define HDIM    128
#define KT      4          // 128 / 32 k-tiles
#define CT_H    8          // 128 / 16 hidden col-tiles
#define CT_ALL  24         // 384 / 16 gate col-tiles
#define FRAG_ELEMS 16      // bf16 per lane per B fragment (8 VGPRs)
#define FRAG_STRIDE (32 * FRAG_ELEMS)          // elements per fragment (all lanes)
#define W_FRAGS  (CT_ALL * KT)                 // 96 fragments per weight matrix
#define W_ELEMS  (W_FRAGS * FRAG_STRIDE)       // 49152 bf16 per weight matrix
#define WAVES    3         // one wave per gate group (r, z, n)
#define HPAD     132       // pad LDS rows to dodge bank conflicts

// A-matrix (16x32 bf16) per-lane k pattern (ISA 7.12.2): element e in v16bf,
// hi = lane>=16. Pairs p=e/2: VGPR0..3 -> K 0..7 (+hi*8), VGPR4..7 -> K 16..23 (+hi*8)
__device__ __forceinline__ int a_klocal(int e, int hi) {
  int p = e >> 1;
  return ((p < 4) ? 0 : 16) + hi * 8 + ((p & 3) << 1) + (e & 1);
}
// B-matrix (32x16 bf16): same per-lane k pattern as the A-matrix
__device__ __forceinline__ int b_klocal(int e, int hi) { return a_klocal(e, hi); }

__device__ __forceinline__ float sigmoidf_(float x) {
  return 1.0f / (1.0f + __expf(-x));
}

// One-thread-per-element weight swizzle: f32 [384][128] -> bf16 B fragments in d_ws.
__global__ void swizzle_w_kernel(const float* __restrict__ w_ih,
                                 const float* __restrict__ w_hh,
                                 __bf16* __restrict__ ws) {
  int tid = blockIdx.x * 256 + threadIdx.x;
  if (tid >= 2 * W_ELEMS) return;
  int w      = tid / W_ELEMS;
  int r      = tid % W_ELEMS;
  int frag   = r / FRAG_STRIDE;          // 0..95  (ct*KT + kt)
  int within = r % FRAG_STRIDE;
  int lane   = within / FRAG_ELEMS;      // 0..31
  int e      = within % FRAG_ELEMS;      // 0..15
  int ct = frag / KT, kt = frag % KT;
  int g = ct * 16 + (lane & 15);                 // gate/output column (0..383)
  int k = kt * 32 + b_klocal(e, lane >> 4);      // reduction index (0..127)
  const float* src = w ? w_hh : w_ih;
  ws[tid] = (__bf16)src[g * IDIM + k];
}

__global__ __launch_bounds__(WAVES * 32)
void gru_scan_kernel(const float* __restrict__ x,
                     const float* __restrict__ hxs,
                     const float* __restrict__ masks,
                     const float* __restrict__ b_ih,
                     const float* __restrict__ b_hh,
                     const __bf16* __restrict__ wsw,
                     float* __restrict__ out) {
  __shared__ float lds_h[16][HPAD];          // h state, f32 C-style [row][col]
  __shared__ float pre_i[3][16][HPAD];       // x-side pre-activations per gate
  __shared__ float pre_h[3][16][HPAD];       // h-side pre-activations per gate
  __shared__ float lds_m[16];                // per-row reset masks for current t
  __shared__ float lds_cbr[HDIM];            // b_ih_r + b_hh_r   (fused)
  __shared__ float lds_cbz[HDIM];            // b_ih_z + b_hh_z   (fused)
  __shared__ float lds_bni[HDIM];            // b_ih_n
  __shared__ float lds_bnh[HDIM];            // b_hh_n (stays inside r*(.))

  const int tid  = threadIdx.x;
  const int wave = tid >> 5;                 // 0=r, 1=z, 2=n gate group
  const int lane = tid & 31;
  const int hi   = lane >> 4;                // half-wave group
  const int lm   = lane & 15;                // M (A rows / C cols) within group
  const int n0   = blockIdx.x * 16;          // this block's batch rows
  const int go   = wave * CT_H;              // gate's ctile offset into 24 ctiles

  const __bf16* wB_ih = wsw;
  const __bf16* wB_hh = wsw + W_ELEMS;

  // init h state and the LDS bias table (block-wide, once)
  for (int idx = tid; idx < 16 * HDIM; idx += WAVES * 32) {
    int rr = idx >> 7, cc = idx & 127;
    lds_h[rr][cc] = hxs[(n0 + rr) * HDIM + cc];
  }
  for (int c = tid; c < HDIM; c += WAVES * 32) {
    lds_cbr[c] = b_ih[c] + b_hh[c];
    lds_cbz[c] = b_ih[128 + c] + b_hh[128 + c];
    lds_bni[c] = b_ih[256 + c];
    lds_bnh[c] = b_hh[256 + c];
  }
  __syncthreads();

  for (int t = 0; t < T_STEPS; ++t) {
    // stage per-row reset masks for this timestep
    if (tid < 16) lds_m[tid] = masks[t * NBATCH + n0 + tid];
    __syncthreads();
    const float maskA = lds_m[lm];

    // ---- build A fragments: x tile (global, prefetched last step) + masked h ----
    const float* xrow = x + ((size_t)(n0 + lm) * T_STEPS + t) * IDIM;
    v16bf a_x[KT], a_h[KT];
    #pragma unroll
    for (int kt = 0; kt < KT; ++kt) {
      v16bf ax, ah;
      #pragma unroll
      for (int e = 0; e < 16; ++e) {
        int k = kt * 32 + a_klocal(e, hi);
        ax[e] = (__bf16)xrow[k];
        ah[e] = (__bf16)(lds_h[lm][k] * maskA);
      }
      a_x[kt] = ax;
      a_h[kt] = ah;
    }

    // ---- zero-register prefetch of next timestep's x rows (global_prefetch_b8) ----
    if (t + 1 < T_STEPS) {
      const char* nx = (const char*)(xrow + IDIM);   // row (n0+lm, t+1), contiguous
      __builtin_prefetch(nx + hi * 256, 0, 1);
      __builtin_prefetch(nx + hi * 256 + 128, 0, 1);
    }

    // ---- this wave's gate GEMMs: 8 ctiles x (4 + 4) WMMAs ----
    #pragma unroll 2
    for (int ct = 0; ct < CT_H; ++ct) {
      v8f acc_i = {};   // x . W_ih[gate]
      v8f acc_h = {};   // h . W_hh[gate]
      #pragma unroll
      for (int kt = 0; kt < KT; ++kt) {
        const v16bf bi = *(const v16bf*)(wB_ih + (((go + ct) * KT + kt) * 32 + lane) * FRAG_ELEMS);
        const v16bf bh = *(const v16bf*)(wB_hh + (((go + ct) * KT + kt) * 32 + lane) * FRAG_ELEMS);
        acc_i = __builtin_amdgcn_wmma_f32_16x16x32_bf16(false, a_x[kt], false, bi, (short)0, acc_i, false, false);
        acc_h = __builtin_amdgcn_wmma_f32_16x16x32_bf16(false, a_h[kt], false, bh, (short)0, acc_h, false, false);
      }
      const int c = ct * 16 + lm;
      #pragma unroll
      for (int v = 0; v < 8; ++v) {
        const int m = v + hi * 8;          // C layout: VGPR v -> row M = v + 8*hi
        pre_i[wave][m][c] = acc_i[v];
        pre_h[wave][m][c] = acc_h[v];
      }
    }

    __syncthreads();   // pre-activations visible; A-frag reads of lds_h complete

    // ---- elementwise GRU update, col-tiles split across the 3 waves ----
    for (int ct = wave; ct < CT_H; ct += WAVES) {
      const int c = ct * 16 + lm;
      const float cbr = lds_cbr[c], cbz = lds_cbz[c];
      const float bni = lds_bni[c], bnh = lds_bnh[c];
      #pragma unroll
      for (int v = 0; v < 8; ++v) {
        const int m = v + hi * 8;
        const float hprev = lds_h[m][c] * lds_m[m];                 // masked h_{t-1}
        const float r_g  = sigmoidf_(pre_i[0][m][c] + pre_h[0][m][c] + cbr);
        const float z_g  = sigmoidf_(pre_i[1][m][c] + pre_h[1][m][c] + cbz);
        const float cand = tanhf((pre_i[2][m][c] + bni) + r_g * (pre_h[2][m][c] + bnh));
        const float hnew = (1.0f - z_g) * cand + z_g * hprev;
        out[((size_t)(n0 + m) * T_STEPS + t) * HDIM + c] = hnew;
        lds_h[m][c] = hnew;                                         // state for next step
      }
    }
    __syncthreads();   // h / mask consumers done before next step's overwrites
  }

  // h_last appended after x_out
  float* hlast = out + (size_t)T_STEPS * NBATCH * HDIM;
  for (int idx = tid; idx < 16 * HDIM; idx += WAVES * 32) {
    int rr = idx >> 7, cc = idx & 127;
    hlast[(n0 + rr) * HDIM + cc] = lds_h[rr][cc];
  }
}

extern "C" void kernel_launch(void* const* d_in, const int* in_sizes, int n_in,
                              void* d_out, int out_size, void* d_ws, size_t ws_size,
                              hipStream_t stream) {
  const float* x     = (const float*)d_in[0];
  const float* hxs   = (const float*)d_in[1];
  const float* masks = (const float*)d_in[2];
  const float* w_ih  = (const float*)d_in[3];
  const float* w_hh  = (const float*)d_in[4];
  const float* b_ih  = (const float*)d_in[5];
  const float* b_hh  = (const float*)d_in[6];
  __bf16* wsw = (__bf16*)d_ws;   // needs 2*W_ELEMS*2 = 192 KB scratch

  swizzle_w_kernel<<<(2 * W_ELEMS + 255) / 256, 256, 0, stream>>>(w_ih, w_hh, wsw);
  gru_scan_kernel<<<NBATCH / 16, WAVES * 32, 0, stream>>>(
      x, hxs, masks, b_ih, b_hh, wsw, (float*)d_out);
}
